// MultiHeadLinearAttentionLayer_31473520345439
// MI455X (gfx1250) — matrix operand order repaired and checked by reading, not compiled, not run on hardware
//
#include <hip/hip_runtime.h>

// Problem constants
#define BB     4
#define SS     2048
#define DMODEL 1024
#define NH     16
#define DH     64
#define NF     64

typedef __attribute__((ext_vector_type(16))) _Float16 v16h;
typedef __attribute__((ext_vector_type(8)))  _Float16 v8h;
typedef __attribute__((ext_vector_type(8)))  float    v8f;
typedef int v4i_vs __attribute__((vector_size(16)));

typedef __attribute__((address_space(1))) v4i_vs* gptr_v4i;
typedef __attribute__((address_space(3))) v4i_vs* lptr_v4i;

__device__ __forceinline__ v8f zero8() {
    v8f z;
#pragma unroll
    for (int i = 0; i < 8; ++i) z[i] = 0.0f;
    return z;
}

__device__ __forceinline__ v8f wmma_f16(v16h a, v16h b, v8f c) {
    // D = A(16x32 f16) * B(32x16 f16) + C(16x16 f32)
    return __builtin_amdgcn_wmma_f32_16x16x32_f16(
        false, a, false, b, (short)0, c, false, false);
}

// ---- CDNA5 async global->LDS (ASYNCcnt-tracked), 16B per lane ----
__device__ __forceinline__ void async_ld16(_Float16* ldst,
                                           const _Float16* gsrc) {
#if defined(__has_builtin) && __has_builtin(__builtin_amdgcn_global_load_async_to_lds_b128)
    __builtin_amdgcn_global_load_async_to_lds_b128(
        (gptr_v4i)(void*)(size_t)gsrc,
        (lptr_v4i)(void*)ldst, 0, 0);
#else
    unsigned lofs = (unsigned)(size_t)(__attribute__((address_space(3))) void*)(void*)ldst;
    unsigned long long ga = (unsigned long long)(size_t)gsrc;
    asm volatile("global_load_async_to_lds_b128 %0, %1, off"
                 :: "v"(lofs), "v"(ga) : "memory");
#endif
}

template <int N>
__device__ __forceinline__ void wait_asynccnt() {
#if defined(__has_builtin) && __has_builtin(__builtin_amdgcn_s_wait_asynccnt)
    __builtin_amdgcn_s_wait_asynccnt(N);
#else
    asm volatile("s_wait_asynccnt %0" :: "i"(N) : "memory");
#endif
}

// A-fragment (16x32 f16) from row-major f16 tile. Lane l: row = l&15,
// elems 0..7 = K[k0+8*(l>>4) .. +7], elems 8..15 = K[k0+16+8*(l>>4) .. +7].
__device__ __forceinline__ v16h load_a_frag(const _Float16* tile, int stride,
                                            int k0, int lane) {
    int m  = lane & 15;
    int hf = (lane >> 4) & 1;
    const _Float16* p = tile + m * stride + k0 + 8 * hf;
    v16h a;
#pragma unroll
    for (int i = 0; i < 8; ++i) a[i] = p[i];
#pragma unroll
    for (int i = 0; i < 8; ++i) a[8 + i] = p[16 + i];
    return a;
}

// Same but source is f32 (global inputs), converted on the fly.
__device__ __forceinline__ v16h load_a_frag_f32(const float* tile, int stride,
                                                int k0, int lane) {
    int m  = lane & 15;
    int hf = (lane >> 4) & 1;
    const float* p = tile + (size_t)m * stride + k0 + 8 * hf;
    v16h a;
#pragma unroll
    for (int i = 0; i < 8; ++i) a[i] = (_Float16)p[i];
#pragma unroll
    for (int i = 0; i < 8; ++i) a[8 + i] = (_Float16)p[16 + i];
    return a;
}

// B-fragment (32x16 f16) from row-major [K][N] f16 tile. Lane l holds column
// N = n0 + (l&15); elems 0..7 = K[k0+8*(l>>4) .. +7], elems 8..15 = K[k0+16+8*(l>>4) .. +7].
__device__ __forceinline__ v16h load_b_frag(const _Float16* tile, int stride,
                                            int k0, int n0, int lane) {
    const _Float16* p = tile + (k0 + 8 * ((lane >> 4) & 1)) * stride + n0 + (lane & 15);
    v16h b;
#pragma unroll
    for (int i = 0; i < 8; ++i) b[i] = p[i * stride];
#pragma unroll
    for (int i = 0; i < 8; ++i) b[8 + i] = p[(16 + i) * stride];
    return b;
}

// 16x64 = A(16x64) @ W(64x64), W in LDS f16 row-major stride 64.
__device__ __forceinline__ void mm64(v16h a0, v16h a1, const _Float16* w,
                                     int lane, v8f c[4]) {
#pragma unroll
    for (int n = 0; n < 4; ++n) {
        c[n] = wmma_f16(a0, load_b_frag(w, DH, 0,  n * 16, lane), c[n]);
        c[n] = wmma_f16(a1, load_b_frag(w, DH, 32, n * 16, lane), c[n]);
    }
}

// C fragments (16x64 f32) + bias [-> cos] -> f16 scratch [16][72]
__device__ __forceinline__ void store_c_scratch(_Float16* sc, const v8f c[4],
                                                const float* bias, bool do_cos,
                                                int lane) {
    int n0 = lane & 15;
    int hf = (lane >> 4) & 1;
#pragma unroll
    for (int n = 0; n < 4; ++n) {
        float bb = bias[n * 16 + n0];
#pragma unroll
        for (int r = 0; r < 8; ++r) {
            float v = c[n][r] + bb;
            if (do_cos) v = __cosf(v);
            sc[(r + 8 * hf) * 72 + n * 16 + n0] = (_Float16)v;
        }
    }
}

// Copy scratch tile [16][72] -> global row-major [16][dstride]
__device__ __forceinline__ void scratch_to_global(const _Float16* sc,
                                                  _Float16* dst, int dstride,
                                                  int lane) {
    int r = lane >> 1, part = lane & 1;
    const v8h* s = (const v8h*)(sc + r * 72 + part * 32);
    v8h* d = (v8h*)(dst + (size_t)r * dstride + part * 32);
#pragma unroll
    for (int i = 0; i < 4; ++i) d[i] = s[i];
}

// ------------------------------------------------------------------
// Kernel 1: per-head projections + cos feature maps.
// Emits Qf [B,H,S,F] f16, KfT [B,H,F,S] f16 (transposed), Vp [B,H,S,D] f16.
// ------------------------------------------------------------------
__global__ __launch_bounds__(256) void proj_features_kernel(
    const float* __restrict__ query, const float* __restrict__ keyin,
    const float* __restrict__ Wq, const float* __restrict__ bq,
    const float* __restrict__ Wk, const float* __restrict__ bk,
    const float* __restrict__ Wv, const float* __restrict__ bv,
    const float* __restrict__ rw, const float* __restrict__ rb,
    _Float16* __restrict__ Qf, _Float16* __restrict__ KfT,
    _Float16* __restrict__ Vp) {
    __shared__ _Float16 wq_s[DH * DH];
    __shared__ _Float16 wk_s[DH * DH];
    __shared__ _Float16 wv_s[DH * DH];
    __shared__ _Float16 rw_s[DH * NF];
    __shared__ float bq_s[DH], bk_s[DH], bv_s[DH], rb_s[NF];
    __shared__ _Float16 sc_s[8][16 * 72];   // per-wave 16x64 staging

    const int bx   = blockIdx.x;
    const int sblk = bx & 15;          // S/128 = 16
    const int h    = (bx >> 4) & 15;
    const int b    = bx >> 8;
    const int tid  = threadIdx.x;
    const int wave = tid >> 5;
    const int lane = tid & 31;
    const int m    = lane & 15;
    const int hf   = (lane >> 4) & 1;
    const int srow0 = sblk * 128 + wave * 16;
    const size_t bh = (size_t)b * NH + h;

#pragma unroll
    for (int i = 0; i < 16; ++i) {
        int idx = tid + 256 * i;
        wq_s[idx] = (_Float16)Wq[(size_t)h * 4096 + idx];
        wk_s[idx] = (_Float16)Wk[(size_t)h * 4096 + idx];
        wv_s[idx] = (_Float16)Wv[(size_t)h * 4096 + idx];
        rw_s[idx] = (_Float16)rw[(size_t)h * 4096 + idx];
    }
    if (tid < 64) {
        bq_s[tid] = bq[h * 64 + tid];
        bk_s[tid] = bk[h * 64 + tid];
        bv_s[tid] = bv[h * 64 + tid];
        rb_s[tid] = rb[h * 64 + tid];
    }
    __syncthreads();

    _Float16* sc = &sc_s[wave][0];
    const float* qtile = query + ((size_t)b * SS + srow0) * DMODEL + h * DH;
    const float* ktile = keyin + ((size_t)b * SS + srow0) * DMODEL + h * DH;

    // ---- Q path: (qh @ Wq + bq) -> cos(. @ rw + rb) ----
    {
        v16h a0 = load_a_frag_f32(qtile, DMODEL, 0, lane);
        v16h a1 = load_a_frag_f32(qtile, DMODEL, 32, lane);
        v8f c[4] = {zero8(), zero8(), zero8(), zero8()};
        mm64(a0, a1, wq_s, lane, c);
        store_c_scratch(sc, c, bq_s, false, lane);
        asm volatile("s_wait_dscnt 0x0" ::: "memory");
        v16h q0 = load_a_frag(sc, 72, 0, lane);
        v16h q1 = load_a_frag(sc, 72, 32, lane);
        v8f f[4] = {zero8(), zero8(), zero8(), zero8()};
        mm64(q0, q1, rw_s, lane, f);
        asm volatile("s_wait_dscnt 0x0" ::: "memory");
        store_c_scratch(sc, f, rb_s, true, lane);
        asm volatile("s_wait_dscnt 0x0" ::: "memory");
        scratch_to_global(sc, Qf + (bh * SS + srow0) * NF, NF, lane);
    }

    // ---- K / V paths share the kh input fragments ----
    v16h ia0 = load_a_frag_f32(ktile, DMODEL, 0, lane);
    v16h ia1 = load_a_frag_f32(ktile, DMODEL, 32, lane);
    {
        v8f c[4] = {zero8(), zero8(), zero8(), zero8()};
        mm64(ia0, ia1, wk_s, lane, c);
        asm volatile("s_wait_dscnt 0x0" ::: "memory");
        store_c_scratch(sc, c, bk_s, false, lane);
        asm volatile("s_wait_dscnt 0x0" ::: "memory");
        v16h k0 = load_a_frag(sc, 72, 0, lane);
        v16h k1 = load_a_frag(sc, 72, 32, lane);
        v8f f[4] = {zero8(), zero8(), zero8(), zero8()};
        mm64(k0, k1, rw_s, lane, f);
        // Write Kf transposed: element [f=n][s=srow0+r+8*hf]; r-consecutive
        // values are s-contiguous -> one 16B store per n-chunk per lane.
        _Float16* kft = KfT + bh * (size_t)NF * SS;
#pragma unroll
        for (int n = 0; n < 4; ++n) {
            int fidx = n * 16 + m;
            float bb = rb_s[fidx];
            v8h pack;
#pragma unroll
            for (int r = 0; r < 8; ++r) pack[r] = (_Float16)__cosf(f[n][r] + bb);
            *(v8h*)(kft + (size_t)fidx * SS + srow0 + 8 * hf) = pack;
        }
    }
    {
        v8f c[4] = {zero8(), zero8(), zero8(), zero8()};
        mm64(ia0, ia1, wv_s, lane, c);
        asm volatile("s_wait_dscnt 0x0" ::: "memory");
        store_c_scratch(sc, c, bv_s, false, lane);
        asm volatile("s_wait_dscnt 0x0" ::: "memory");
        scratch_to_global(sc, Vp + (bh * SS + srow0) * DH, DH, lane);
    }
}

// ------------------------------------------------------------------
// Kernel 2: flash attention, 64-key tiles, double-buffered async staging.
// scores = Qf·Kfᵀ, online softmax, out = P·V.
// Writes merged [B*S, DMODEL] f16 (head-interleaved layout for final GEMM).
// ------------------------------------------------------------------
__global__ __launch_bounds__(256) void flash_attn_kernel(
    const _Float16* __restrict__ Qf, const _Float16* __restrict__ KfT,
    const _Float16* __restrict__ Vp, _Float16* __restrict__ merged) {
    __shared__ _Float16 kft_s[2][64 * 72];   // [buf][f][t]  stride 72
    __shared__ _Float16 v_s[2][64 * 72];     // [buf][t][d]  stride 72
    __shared__ _Float16 p_s[8][16 * 72];     // per-wave P 16x64, stride 72

    const int bx   = blockIdx.x;
    const int sblk = bx & 15;
    const int h    = (bx >> 4) & 15;
    const int b    = bx >> 8;
    const int tid  = threadIdx.x;
    const int wave = tid >> 5;
    const int lane = tid & 31;
    const int m    = lane & 15;
    const int hf   = (lane >> 4) & 1;
    const int srow0 = sblk * 128 + wave * 16;
    const size_t bh = (size_t)b * NH + h;

    // Q fragments held in registers for the whole K loop.
    const _Float16* qtile = Qf + (bh * SS + srow0) * NF;
    const v16h qa0 = load_a_frag(qtile, NF, 0, lane);
    const v16h qa1 = load_a_frag(qtile, NF, 32, lane);

    float mrow[8], lrow[8];
#pragma unroll
    for (int r = 0; r < 8; ++r) { mrow[r] = -1.0e30f; lrow[r] = 0.0f; }
    v8f acc[4] = {zero8(), zero8(), zero8(), zero8()};

    const _Float16* kft_g = KfT + bh * (size_t)NF * SS;
    const _Float16* v_g   = Vp + bh * (size_t)SS * DH;
    _Float16* pw = &p_s[wave][0];

    // async staging: each thread DMAs 4x16B per tile (2 Kfᵀ rows + 2 V rows)
    const int fr = tid >> 3;   // 0..31
    const int ch = tid & 7;    // 0..7
    auto stage = [&](int t0, int buf) {
        async_ld16(&kft_s[buf][fr * 72 + ch * 8],
                   kft_g + (size_t)fr * SS + t0 + ch * 8);
        async_ld16(&kft_s[buf][(fr + 32) * 72 + ch * 8],
                   kft_g + (size_t)(fr + 32) * SS + t0 + ch * 8);
        async_ld16(&v_s[buf][fr * 72 + ch * 8],
                   v_g + (size_t)(t0 + fr) * DH + ch * 8);
        async_ld16(&v_s[buf][(fr + 32) * 72 + ch * 8],
                   v_g + (size_t)(t0 + fr + 32) * DH + ch * 8);
    };

    stage(0, 0);
    const int NT = SS / 64;   // 32 tiles
    for (int it = 0; it < NT; ++it) {
        const int cur = it & 1;
        __syncthreads();                    // buf[cur^1] reads (iter-1) done
        if (it + 1 < NT) {
            stage((it + 1) * 64, cur ^ 1);  // prefetch next tile
            wait_asynccnt<4>();             // tile `it` resident (in-order)
        } else {
            wait_asynccnt<0>();
        }
        __syncthreads();                    // all waves' DMA visible

        const _Float16* kb = &kft_s[cur][0];
        const _Float16* vb = &v_s[cur][0];

        // scores: 16 rows x 64 keys (four 16x16 C tiles), K-dim F=64
        v8f s[4];
#pragma unroll
        for (int n = 0; n < 4; ++n) {
            s[n] = zero8();
            s[n] = wmma_f16(qa0, load_b_frag(kb, 72, 0,  n * 16, lane), s[n]);
            s[n] = wmma_f16(qa1, load_b_frag(kb, 72, 32, n * 16, lane), s[n]);
        }

        // online softmax per row (row = r + 8*hf; cols striped over 16 lanes)
#pragma unroll
        for (int r = 0; r < 8; ++r) {
            float tmax = fmaxf(fmaxf(s[0][r], s[1][r]),
                               fmaxf(s[2][r], s[3][r]));
            tmax = fmaxf(tmax, __shfl_xor(tmax, 1));
            tmax = fmaxf(tmax, __shfl_xor(tmax, 2));
            tmax = fmaxf(tmax, __shfl_xor(tmax, 4));
            tmax = fmaxf(tmax, __shfl_xor(tmax, 8));
            float mnew  = fmaxf(mrow[r], tmax);
            float scale = __expf(mrow[r] - mnew);
            float p[4], psum = 0.0f;
#pragma unroll
            for (int n = 0; n < 4; ++n) {
                p[n] = __expf(s[n][r] - mnew);
                psum += p[n];
            }
            psum += __shfl_xor(psum, 1);
            psum += __shfl_xor(psum, 2);
            psum += __shfl_xor(psum, 4);
            psum += __shfl_xor(psum, 8);
            lrow[r] = lrow[r] * scale + psum;
            mrow[r] = mnew;
#pragma unroll
            for (int n = 0; n < 4; ++n) acc[n][r] *= scale;
            _Float16* prow = pw + (r + 8 * hf) * 72;
#pragma unroll
            for (int n = 0; n < 4; ++n) prow[n * 16 + m] = (_Float16)p[n];
        }
        asm volatile("s_wait_dscnt 0x0" ::: "memory");

        // out += P(16x64) @ V(64x64)
        v16h pa0 = load_a_frag(pw, 72, 0, lane);
        v16h pa1 = load_a_frag(pw, 72, 32, lane);
#pragma unroll
        for (int n = 0; n < 4; ++n) {
            acc[n] = wmma_f16(pa0, load_b_frag(vb, 72, 0,  n * 16, lane), acc[n]);
            acc[n] = wmma_f16(pa1, load_b_frag(vb, 72, 32, n * 16, lane), acc[n]);
        }
    }

    // epilogue: normalize, write merged [b*S + s][h*64 + d]
#pragma unroll
    for (int n = 0; n < 4; ++n) {
#pragma unroll
        for (int r = 0; r < 8; ++r) {
            float o = acc[n][r] / lrow[r];
            int srow = srow0 + r + 8 * hf;
            merged[((size_t)b * SS + srow) * DMODEL + h * DH + n * 16 + m] =
                (_Float16)o;
        }
    }
}

// ------------------------------------------------------------------
// Kernel 3: out = merged(8192x1024 f16) @ Wo(1024x1024 f32->f16) + bo, f32 out
// Block tile 128(M) x 64(N), k-step 32.  8 waves: 16 rows x 64 cols each.
// ------------------------------------------------------------------
__global__ __launch_bounds__(256) void out_gemm_kernel(
    const _Float16* __restrict__ merged, const float* __restrict__ Wo,
    const float* __restrict__ bo, float* __restrict__ out) {
    __shared__ _Float16 wo_s[32 * 72];      // [k][n] stride 72

    const int bx   = blockIdx.x;
    const int n0   = (bx & 15) * 64;        // N/64 = 16
    const int m0   = (bx >> 4) * 128;       // M/128 = 64
    const int tid  = threadIdx.x;
    const int wave = tid >> 5;
    const int lane = tid & 31;
    const int m    = lane & 15;
    const int hf   = (lane >> 4) & 1;
    const int mrow0 = m0 + wave * 16;

    v8f c[4] = {zero8(), zero8(), zero8(), zero8()};
    const _Float16* atile = merged + (size_t)mrow0 * DMODEL;

    for (int k0 = 0; k0 < DMODEL; k0 += 32) {
        __syncthreads();
        {   // stage Wo tile 32x64, f32 -> f16
            int k = tid >> 3, ch = tid & 7;
            const float* src = Wo + (size_t)(k0 + k) * DMODEL + n0 + ch * 8;
            v8h pk;
#pragma unroll
            for (int i = 0; i < 8; ++i) pk[i] = (_Float16)src[i];
            *(v8h*)&wo_s[k * 72 + ch * 8] = pk;
        }
        __syncthreads();

        v16h a = load_a_frag(atile, DMODEL, k0, lane);
#pragma unroll
        for (int n = 0; n < 4; ++n)
            c[n] = wmma_f16(a, load_b_frag(wo_s, 72, 0, n * 16, lane), c[n]);
    }

#pragma unroll
    for (int n = 0; n < 4; ++n) {
        float bb = bo[n0 + n * 16 + m];
#pragma unroll
        for (int r = 0; r < 8; ++r) {
            out[(size_t)(mrow0 + r + 8 * hf) * DMODEL + n0 + n * 16 + m] =
                c[n][r] + bb;
        }
    }
}

// ------------------------------------------------------------------
extern "C" void kernel_launch(void* const* d_in, const int* in_sizes, int n_in,
                              void* d_out, int out_size, void* d_ws,
                              size_t ws_size, hipStream_t stream) {
    const float* query = (const float*)d_in[0];
    const float* keyin = (const float*)d_in[1];
    const float* Wq = (const float*)d_in[2];
    const float* bq = (const float*)d_in[3];
    const float* Wk = (const float*)d_in[4];
    const float* bk = (const float*)d_in[5];
    const float* Wv = (const float*)d_in[6];
    const float* bv = (const float*)d_in[7];
    const float* rw = (const float*)d_in[8];
    const float* rb = (const float*)d_in[9];
    const float* Wo = (const float*)d_in[10];
    const float* bo = (const float*)d_in[11];
    float* out = (float*)d_out;

    _Float16* ws = (_Float16*)d_ws;
    const size_t nF = (size_t)BB * NH * SS * NF;     // 8,388,608 halfs each
    _Float16* Qf     = ws;
    _Float16* KfT    = Qf + nF;
    _Float16* Vp     = KfT + nF;
    _Float16* merged = Vp + nF;                      // B*S*DMODEL halfs

    const int nblk = BB * NH * (SS / 128);           // 1024
    proj_features_kernel<<<dim3(nblk), dim3(256), 0, stream>>>(
        query, keyin, Wq, bq, Wk, bk, Wv, bv, rw, rb, Qf, KfT, Vp);
    flash_attn_kernel<<<dim3(nblk), dim3(256), 0, stream>>>(Qf, KfT, Vp,
                                                            merged);
    const int gblk = (BB * SS / 128) * (DMODEL / 64); // 1024
    out_gemm_kernel<<<dim3(gblk), dim3(256), 0, stream>>>(merged, Wo, bo, out);
}
